// RelPosTransformer_52355651338428
// MI455X (gfx1250) — compile-verified, matrix operand order checked by reading, not run on hardware
//
#include <hip/hip_runtime.h>
#include <math.h>

// ---------------- problem constants ----------------
#define SQ      512          // sequence length
#define BBATCH  4            // batch
#define DMODEL  1024         // model dim
#define NHEAD   8
#define HDIM    128          // head dim
#define DFFN    1024
#define RREL    1023         // 2S-1
#define RPAD    1024         // padded rel rows (row 1023 zeroed)
#define NTOK    2048         // BBATCH*SQ
#define NBH     32           // BBATCH*NHEAD
#define ATT_SCALE 0.08838834764831845f   // 1/sqrt(128)

#define BROW    40           // LDS row stride (32 K elems + 8 pad), 80B (16B-aligned)

typedef __attribute__((ext_vector_type(16))) __bf16 v16bf;
typedef __attribute__((ext_vector_type(8)))  float  v8f;

union Frag { v16bf v; uint4 u[2]; };

#if __has_builtin(__builtin_amdgcn_global_load_async_to_lds_b128)
#define HAVE_ASYNC_LDS 1
#else
#define HAVE_ASYNC_LDS 0
#endif

#if HAVE_ASYNC_LDS
typedef int v4ivs __attribute__((vector_size(4 * sizeof(int))));
typedef __attribute__((address_space(1))) void* as1_void;
typedef __attribute__((address_space(3))) void* as3_void;
typedef __attribute__((address_space(1))) v4ivs* as1_v4i;
typedef __attribute__((address_space(3))) v4ivs* as3_v4i;
#endif

__device__ __forceinline__ unsigned short f2bf(float x) {
  unsigned int u = __float_as_uint(x);
  u += 0x7FFFu + ((u >> 16) & 1u);        // round-to-nearest-even
  return (unsigned short)(u >> 16);
}

// 16-byte global -> LDS copy: async DMA on CDNA5, sync fallback otherwise.
__device__ __forceinline__ void cp16_g2l(const unsigned short* g, unsigned short* l) {
#if HAVE_ASYNC_LDS
  __builtin_amdgcn_global_load_async_to_lds_b128(
      (as1_v4i)(as1_void)(void*)g,
      (as3_v4i)(as3_void)(void*)l, 0, 0);
#else
  *(uint4*)l = *(const uint4*)g;
#endif
}

__device__ __forceinline__ void async_wait0() {
#if HAVE_ASYNC_LDS
#if __has_builtin(__builtin_amdgcn_s_wait_asynccnt)
  __builtin_amdgcn_s_wait_asynccnt(0);
#else
  asm volatile("s_wait_asynccnt 0x0" ::: "memory");
#endif
#endif
}

// =====================================================================
// Generic bf16 WMMA GEMM:  C[M,N] = A[M,K] * Bt[N,K]^T  (+bias +resid, GELU)
// A rows stream straight from global (2x uint4 per 16x16x32 fragment).
// The 64x32 B tile (shared by all 4 waves) is staged into LDS with
// GLOBAL_LOAD_ASYNC_TO_LDS_B128 and double-buffered so the copy of
// K-step k+32 overlaps the WMMAs of K-step k; fragments then come from
// ds_load_b128.  Batched: per-z offsets = (z/8)*s1 + (z%8)*s2.
// Block: 128 threads = 4 waves; 64x64 tile; wave w -> rows [16w,16w+16).
// M,N multiples of 64 and K multiple of 32 guaranteed by the caller.
// =====================================================================
__global__ __launch_bounds__(128) void gemm_wmma_kernel(
    const unsigned short* __restrict__ A,  long lda, long sA1, long sA2,
    const unsigned short* __restrict__ Bt, long ldb, long sB1, long sB2,
    float* __restrict__ C, unsigned short* __restrict__ Cb,
    long ldc, long sC1, long sC2,
    const float* __restrict__ bias, const float* __restrict__ resid,
    int M, int N, int K, int gelu)
{
  __shared__ unsigned short tileB[2][64 * BROW];

  long zq = (long)(blockIdx.z >> 3), zr = (long)(blockIdx.z & 7);
  const unsigned short* Ab = A  + zq * sA1 + zr * sA2;
  const unsigned short* Bb = Bt + zq * sB1 + zr * sB2;
  long offC = zq * sC1 + zr * sC2;

  int tid  = (int)threadIdx.x;
  int wave = tid >> 5;
  int lane = tid & 31;
  int l  = lane & 15;          // row (A) / col (B,C)
  int hi = lane >> 4;          // K half-select

  int rowBase = (int)blockIdx.y * 64 + wave * 16;
  int colBase = (int)blockIdx.x * 64;

  // B-tile copy assignment: 4KB/step = 256 x 16B chunks, 2 per thread.
  int ci0 = tid * 2, ci1 = tid * 2 + 1;
  int r0 = ci0 >> 2, o0 = (ci0 & 3) * 8;   // element offsets
  int r1 = ci1 >> 2, o1 = (ci1 & 3) * 8;
  const unsigned short* bsrc0 = Bb + (long)(colBase + r0) * ldb + o0;
  const unsigned short* bsrc1 = Bb + (long)(colBase + r1) * ldb + o1;
  int ldst0 = r0 * BROW + o0;
  int ldst1 = r1 * BROW + o1;

  v8f acc[4] = {v8f{}, v8f{}, v8f{}, v8f{}};

  const unsigned short* ap = Ab + (long)(rowBase + l) * lda + hi * 8;

  // prologue: stage first B tile
  cp16_g2l(bsrc0, &tileB[0][ldst0]);
  cp16_g2l(bsrc1, &tileB[0][ldst1]);
  int cur = 0;

  for (int k0 = 0; k0 < K; k0 += 32) {
    async_wait0();          // this wave's async copies into tileB[cur] done
    __syncthreads();        // all waves' copies visible

    if (k0 + 32 < K) {      // overlap next tile's copy with this tile's math
      cp16_g2l(bsrc0 + k0 + 32, &tileB[cur ^ 1][ldst0]);
      cp16_g2l(bsrc1 + k0 + 32, &tileB[cur ^ 1][ldst1]);
    }

    Frag a;
    a.u[0] = *(const uint4*)(ap + k0);
    a.u[1] = *(const uint4*)(ap + k0 + 16);
    __builtin_prefetch(ap + k0 + 32, 0, 0);

#pragma unroll
    for (int t = 0; t < 4; ++t) {
      const unsigned short* bl = &tileB[cur][(t * 16 + l) * BROW + hi * 8];
      Frag b;
      b.u[0] = *(const uint4*)(bl);
      b.u[1] = *(const uint4*)(bl + 16);
      acc[t] = __builtin_amdgcn_wmma_f32_16x16x32_bf16(
          false, a.v, false, b.v, (short)0, acc[t], false, false);
    }

    __syncthreads();        // done reading tileB[cur] before it is re-filled
    cur ^= 1;
  }

#pragma unroll
  for (int t = 0; t < 4; ++t) {
    int col = colBase + t * 16 + l;
    float bv = bias ? bias[col] : 0.0f;
#pragma unroll
    for (int r = 0; r < 8; ++r) {
      int row = rowBase + hi * 8 + r;
      long idx = offC + (long)row * ldc + col;
      float v = acc[t][r] + bv;
      if (resid) v += resid[idx];
      if (gelu)  v = 0.5f * v * (1.0f + erff(v * 0.70710678118654752f));
      if (C)  C[idx]  = v;
      if (Cb) Cb[idx] = f2bf(v);
    }
  }
}

// ---------------- elementwise / helper kernels ----------------
__global__ void cvt_bf16_kernel(const float* __restrict__ s,
                                unsigned short* __restrict__ d, long n) {
  long i = (long)blockIdx.x * blockDim.x + threadIdx.x;
  if (i < n) d[i] = f2bf(s[i]);
}

// rel [RREL,HDIM] f32 -> relb [RPAD,HDIM] bf16, last row zero
__global__ void cvt_rel_kernel(const float* __restrict__ s,
                               unsigned short* __restrict__ d) {
  long i = (long)blockIdx.x * blockDim.x + threadIdx.x;
  if (i >= (long)RPAD * HDIM) return;
  long r = i / HDIM;
  d[i] = (r < RREL) ? f2bf(s[i]) : (unsigned short)0;
}

// W[K,N] f32 -> WT[N,K] bf16 via LDS 32x32 tile. block(32,8), grid(N/32,K/32)
__global__ void cvtT_kernel(const float* __restrict__ W,
                            unsigned short* __restrict__ WT, int K, int Nn) {
  __shared__ float t[32][33];
  int tx = threadIdx.x, ty = threadIdx.y;
  int nb = blockIdx.x * 32, kb = blockIdx.y * 32;
#pragma unroll
  for (int r = 0; r < 4; ++r)
    t[ty + r * 8][tx] = W[(long)(kb + ty + r * 8) * Nn + nb + tx];
  __syncthreads();
#pragma unroll
  for (int r = 0; r < 4; ++r)
    WT[(long)(nb + ty + r * 8) * K + kb + tx] = f2bf(t[tx][ty + r * 8]);
}

// vb [NTOK,DMODEL] bf16 -> vT [(bh*HDIM+d)*SQ + j] bf16
__global__ void vtrans_kernel(const unsigned short* __restrict__ vb,
                              unsigned short* __restrict__ vT) {
  long i = (long)blockIdx.x * blockDim.x + threadIdx.x;
  if (i >= (long)NBH * HDIM * SQ) return;
  int j  = (int)(i % SQ);
  long r = i / SQ;
  int d  = (int)(r % HDIM);
  int bh = (int)(r / HDIM);
  int b = bh >> 3, hh = bh & 7;
  vT[i] = vb[(long)(b * SQ + j) * DMODEL + hh * HDIM + d];
}

// av [bh,SQ,HDIM] f32 -> aob [NTOK,DMODEL] bf16 (concat heads)
__global__ void merge_kernel(const float* __restrict__ av,
                             unsigned short* __restrict__ aob) {
  long i = (long)blockIdx.x * blockDim.x + threadIdx.x;
  if (i >= (long)NTOK * DMODEL) return;
  int d2 = (int)(i % DMODEL);
  long row = i / DMODEL;
  int hh = d2 / HDIM, d = d2 % HDIM;
  int b = (int)(row / SQ), s = (int)(row % SQ);
  aob[i] = f2bf(av[(long)((b * NHEAD + hh) * SQ + s) * HDIM + d]);
}

// softmax over j of (scores + qr[i-j+S-1]) * SCALE with optional causal mask
// grid (SQ, NBH), block 128, 4 elems/thread -> bf16 probs
__global__ __launch_bounds__(128) void softmax_kernel(
    const float* __restrict__ scores, const float* __restrict__ qr,
    unsigned short* __restrict__ probs, int causal) {
  int i  = (int)blockIdx.x;
  int bh = (int)blockIdx.y;
  const float* srow = scores + ((long)bh * SQ + i) * SQ;
  const float* qrow = qr     + ((long)bh * SQ + i) * RPAD;
  unsigned short* prow = probs + ((long)bh * SQ + i) * SQ;
  __shared__ float red[128];
  int tid = (int)threadIdx.x;
  float vals[4];
  float m = -3.4e38f;
#pragma unroll
  for (int t = 0; t < 4; ++t) {
    int j = t * 128 + tid;
    float v = (srow[j] + qrow[i - j + (SQ - 1)]) * ATT_SCALE;
    if (causal && j > i) v = -1e9f;
    vals[t] = v;
    m = fmaxf(m, v);
  }
  red[tid] = m; __syncthreads();
  for (int o = 64; o > 0; o >>= 1) {
    if (tid < o) red[tid] = fmaxf(red[tid], red[tid + o]);
    __syncthreads();
  }
  m = red[0]; __syncthreads();
  float sum = 0.0f;
#pragma unroll
  for (int t = 0; t < 4; ++t) { vals[t] = __expf(vals[t] - m); sum += vals[t]; }
  red[tid] = sum; __syncthreads();
  for (int o = 64; o > 0; o >>= 1) {
    if (tid < o) red[tid] += red[tid + o];
    __syncthreads();
  }
  float inv = 1.0f / red[0];
#pragma unroll
  for (int t = 0; t < 4; ++t) prow[t * 128 + tid] = f2bf(vals[t] * inv);
}

// per-row LayerNorm over DMODEL, block 256, 4 elems/thread
__global__ __launch_bounds__(256) void layernorm_kernel(
    const float* __restrict__ in, const float* __restrict__ g,
    const float* __restrict__ b, float* __restrict__ out) {
  long row = (long)blockIdx.x;
  const float* x = in + row * DMODEL;
  float* o = out + row * DMODEL;
  __shared__ float red[256];
  int tid = (int)threadIdx.x;
  float v[4]; float s = 0.0f;
#pragma unroll
  for (int t = 0; t < 4; ++t) { v[t] = x[t * 256 + tid]; s += v[t]; }
  red[tid] = s; __syncthreads();
  for (int off = 128; off > 0; off >>= 1) {
    if (tid < off) red[tid] += red[tid + off];
    __syncthreads();
  }
  float mean = red[0] * (1.0f / DMODEL); __syncthreads();
  float s2 = 0.0f;
#pragma unroll
  for (int t = 0; t < 4; ++t) { float d = v[t] - mean; s2 += d * d; }
  red[tid] = s2; __syncthreads();
  for (int off = 128; off > 0; off >>= 1) {
    if (tid < off) red[tid] += red[tid + off];
    __syncthreads();
  }
  float rs = rsqrtf(red[0] * (1.0f / DMODEL) + 1e-5f);
#pragma unroll
  for (int t = 0; t < 4; ++t) {
    int j = t * 256 + tid;
    o[j] = (v[t] - mean) * rs * g[j] + b[j];
  }
}

// [S,B,D] <-> [B,S,D]
__global__ void tr_sbd_kernel(const float* __restrict__ in,
                              float* __restrict__ out, int toBSD) {
  long i = (long)blockIdx.x * blockDim.x + threadIdx.x;
  if (i >= (long)SQ * BBATCH * DMODEL) return;
  int d = (int)(i % DMODEL);
  long r = i / DMODEL;
  if (toBSD) {
    int b = (int)(r / SQ), s = (int)(r % SQ);
    out[i] = in[(long)(s * BBATCH + b) * DMODEL + d];
  } else {
    int s = (int)(r / BBATCH), b = (int)(r % BBATCH);
    out[i] = in[(long)(b * SQ + s) * DMODEL + d];
  }
}

// ---------------- host-side orchestration ----------------
struct WS {
  float *h, *mem, *tmp, *av, *scores, *qr;
  unsigned short *hb, *memb, *qb, *kb, *vb, *vT, *aob, *mid, *probs, *wt, *relb;
};

static inline void run_gemm(hipStream_t st,
    const unsigned short* A, long lda, long sA1, long sA2,
    const unsigned short* Bt, long ldb, long sB1, long sB2,
    float* C, unsigned short* Cb, long ldc, long sC1, long sC2,
    const float* bias, const float* resid,
    int M, int N, int K, int Z, int gelu) {
  dim3 g((unsigned)(N / 64), (unsigned)(M / 64), (unsigned)Z);
  gemm_wmma_kernel<<<g, 128, 0, st>>>(A, lda, sA1, sA2, Bt, ldb, sB1, sB2,
                                      C, Cb, ldc, sC1, sC2, bias, resid,
                                      M, N, K, gelu);
}

static inline void run_cvt(hipStream_t st, const float* s, unsigned short* d, long n) {
  cvt_bf16_kernel<<<(unsigned)((n + 255) / 256), 256, 0, st>>>(s, d, n);
}

static inline void run_cvtT(hipStream_t st, const float* W, unsigned short* WT,
                            int K, int N) {
  cvtT_kernel<<<dim3((unsigned)(N / 32), (unsigned)(K / 32)), dim3(32, 8), 0, st>>>(W, WT, K, N);
}

// full rel-position attention: tmp = h + Attn(qin, kvin); caller then LNs.
static void run_attention(hipStream_t st, const WS& w,
                          const unsigned short* qin, const unsigned short* kvin,
                          const float* Wqkvo, const float* Bqkvo,
                          const float* relw, int causal) {
  const long DD = DMODEL;
  // Q,K,V projections (bf16 out)
  run_cvtT(st, Wqkvo + 0 * DD * DD, w.wt, DMODEL, DMODEL);
  run_gemm(st, qin, DD, 0, 0, w.wt, DD, 0, 0, nullptr, w.qb, DD, 0, 0,
           Bqkvo + 0 * DD, nullptr, NTOK, DMODEL, DMODEL, 1, 0);
  run_cvtT(st, Wqkvo + 1 * DD * DD, w.wt, DMODEL, DMODEL);
  run_gemm(st, kvin, DD, 0, 0, w.wt, DD, 0, 0, nullptr, w.kb, DD, 0, 0,
           Bqkvo + 1 * DD, nullptr, NTOK, DMODEL, DMODEL, 1, 0);
  run_cvtT(st, Wqkvo + 2 * DD * DD, w.wt, DMODEL, DMODEL);
  run_gemm(st, kvin, DD, 0, 0, w.wt, DD, 0, 0, nullptr, w.vb, DD, 0, 0,
           Bqkvo + 2 * DD, nullptr, NTOK, DMODEL, DMODEL, 1, 0);
  // head-transpose V
  vtrans_kernel<<<(unsigned)(((long)NBH * HDIM * SQ + 255) / 256), 256, 0, st>>>(w.vb, w.vT);
  // rel table -> bf16 (padded)
  cvt_rel_kernel<<<(unsigned)(((long)RPAD * HDIM + 255) / 256), 256, 0, st>>>(relw, w.relb);
  // qr[bh,S,RPAD] = q_bh @ rel^T
  run_gemm(st, w.qb, DD, (long)SQ * DD, HDIM,
           w.relb, HDIM, 0, 0,
           w.qr, nullptr, RPAD, 8L * SQ * RPAD, (long)SQ * RPAD,
           nullptr, nullptr, SQ, RPAD, HDIM, NBH, 0);
  // scores[bh,S,S] = q_bh @ k_bh^T
  run_gemm(st, w.qb, DD, (long)SQ * DD, HDIM,
           w.kb, DD, (long)SQ * DD, HDIM,
           w.scores, nullptr, SQ, 8L * SQ * SQ, (long)SQ * SQ,
           nullptr, nullptr, SQ, SQ, HDIM, NBH, 0);
  // softmax with gathered rel bias -> bf16 probs
  softmax_kernel<<<dim3(SQ, NBH), 128, 0, st>>>(w.scores, w.qr, w.probs, causal);
  // av[bh,S,HDIM] = probs @ v_bh
  run_gemm(st, w.probs, SQ, 8L * SQ * SQ, (long)SQ * SQ,
           w.vT, SQ, 8L * HDIM * SQ, (long)HDIM * SQ,
           w.av, nullptr, HDIM, 8L * SQ * HDIM, (long)SQ * HDIM,
           nullptr, nullptr, SQ, HDIM, SQ, NBH, 0);
  // concat heads -> bf16
  merge_kernel<<<(unsigned)(((long)NTOK * DMODEL + 255) / 256), 256, 0, st>>>(w.av, w.aob);
  // output projection + residual (f32 into tmp)
  run_cvtT(st, Wqkvo + 3 * DD * DD, w.wt, DMODEL, DMODEL);
  run_gemm(st, w.aob, DD, 0, 0, w.wt, DD, 0, 0, w.tmp, nullptr, DD, 0, 0,
           Bqkvo + 3 * DD, w.h, NTOK, DMODEL, DMODEL, 1, 0);
}

static void run_ffn(hipStream_t st, const WS& w,
                    const float* W1, const float* B1,
                    const float* W2, const float* B2) {
  run_cvtT(st, W1, w.wt, DMODEL, DFFN);
  run_gemm(st, w.hb, DMODEL, 0, 0, w.wt, DMODEL, 0, 0, nullptr, w.mid,
           DFFN, 0, 0, B1, nullptr, NTOK, DFFN, DMODEL, 1, /*gelu=*/1);
  run_cvtT(st, W2, w.wt, DFFN, DMODEL);
  run_gemm(st, w.mid, DFFN, 0, 0, w.wt, DFFN, 0, 0, w.tmp, nullptr,
           DMODEL, 0, 0, B2, w.h, NTOK, DMODEL, DFFN, 1, 0);
}

static inline void run_ln(hipStream_t st, const WS& w,
                          const float* g, const float* b) {
  layernorm_kernel<<<NTOK, 256, 0, st>>>(w.tmp, g, b, w.h);
  run_cvt(st, w.h, w.hb, (long)NTOK * DMODEL);
}

extern "C" void kernel_launch(void* const* d_in, const int* in_sizes, int n_in,
                              void* d_out, int out_size, void* d_ws, size_t ws_size,
                              hipStream_t stream) {
  (void)in_sizes; (void)n_in; (void)out_size; (void)ws_size;
  const float* x       = (const float*)d_in[0];
  const float* tgt     = (const float*)d_in[1];
  const float* enc_w   = (const float*)d_in[2];
  const float* enc_b   = (const float*)d_in[3];
  const float* enc_rel = (const float*)d_in[4];
  const float* enc_w1  = (const float*)d_in[5];
  const float* enc_b1  = (const float*)d_in[6];
  const float* enc_w2  = (const float*)d_in[7];
  const float* enc_b2  = (const float*)d_in[8];
  const float* enc_lng = (const float*)d_in[9];
  const float* enc_lnb = (const float*)d_in[10];
  const float* ds_w    = (const float*)d_in[11];
  const float* ds_b    = (const float*)d_in[12];
  const float* ds_rel  = (const float*)d_in[13];
  const float* dc_w    = (const float*)d_in[14];
  const float* dc_b    = (const float*)d_in[15];
  const float* dc_rel  = (const float*)d_in[16];
  const float* dw1     = (const float*)d_in[17];
  const float* db1     = (const float*)d_in[18];
  const float* dw2     = (const float*)d_in[19];
  const float* db2     = (const float*)d_in[20];
  const float* d_lng   = (const float*)d_in[21];
  const float* d_lnb   = (const float*)d_in[22];

  // carve workspace (assumed large enough; ~180 MB total)
  char* p = (char*)d_ws;
  auto alloc = [&](size_t bytes) -> void* {
    void* r = (void*)p;
    p += (bytes + 255) & ~(size_t)255;
    return r;
  };
  WS w;
  w.h      = (float*)alloc((size_t)NTOK * DMODEL * 4);
  w.mem    = (float*)alloc((size_t)NTOK * DMODEL * 4);
  w.tmp    = (float*)alloc((size_t)NTOK * DMODEL * 4);
  w.av     = (float*)alloc((size_t)NBH * SQ * HDIM * 4);
  w.scores = (float*)alloc((size_t)NBH * SQ * SQ * 4);
  w.qr     = (float*)alloc((size_t)NBH * SQ * RPAD * 4);
  w.hb     = (unsigned short*)alloc((size_t)NTOK * DMODEL * 2);
  w.memb   = (unsigned short*)alloc((size_t)NTOK * DMODEL * 2);
  w.qb     = (unsigned short*)alloc((size_t)NTOK * DMODEL * 2);
  w.kb     = (unsigned short*)alloc((size_t)NTOK * DMODEL * 2);
  w.vb     = (unsigned short*)alloc((size_t)NTOK * DMODEL * 2);
  w.vT     = (unsigned short*)alloc((size_t)NBH * HDIM * SQ * 2);
  w.aob    = (unsigned short*)alloc((size_t)NTOK * DMODEL * 2);
  w.mid    = (unsigned short*)alloc((size_t)NTOK * DFFN * 2);
  w.probs  = (unsigned short*)alloc((size_t)NBH * SQ * SQ * 2);
  w.wt     = (unsigned short*)alloc((size_t)DMODEL * DFFN * 2);
  w.relb   = (unsigned short*)alloc((size_t)RPAD * HDIM * 2);

  const long nElem = (long)NTOK * DMODEL;
  unsigned eb = (unsigned)((nElem + 255) / 256);

  // x,[S,B,D] -> h [B,S,D]; tgt -> mem
  tr_sbd_kernel<<<eb, 256, 0, stream>>>(x, w.h, 1);
  run_cvt(stream, w.h, w.hb, nElem);
  tr_sbd_kernel<<<eb, 256, 0, stream>>>(tgt, w.mem, 1);
  run_cvt(stream, w.mem, w.memb, nElem);

  const long WSTRIDE = (long)4 * DMODEL * DMODEL;

  // ---------------- encoder ----------------
  for (int i = 0; i < 6; ++i) {
    run_attention(stream, w, w.hb, w.hb,
                  enc_w + (long)i * WSTRIDE, enc_b + (long)i * 4 * DMODEL,
                  enc_rel + (long)i * RREL * HDIM, /*causal=*/0);
    run_ln(stream, w, enc_lng + (long)(i * 2 + 0) * DMODEL,
                      enc_lnb + (long)(i * 2 + 0) * DMODEL);
    run_ffn(stream, w, enc_w1 + (long)i * DMODEL * DFFN, enc_b1 + (long)i * DFFN,
                      enc_w2 + (long)i * DFFN * DMODEL, enc_b2 + (long)i * DMODEL);
    run_ln(stream, w, enc_lng + (long)(i * 2 + 1) * DMODEL,
                      enc_lnb + (long)(i * 2 + 1) * DMODEL);
  }

  // ---------------- decoder ----------------
  for (int i = 0; i < 6; ++i) {
    run_attention(stream, w, w.hb, w.hb,
                  ds_w + (long)i * WSTRIDE, ds_b + (long)i * 4 * DMODEL,
                  ds_rel + (long)i * RREL * HDIM, /*causal=*/1);
    run_ln(stream, w, d_lng + (long)(i * 3 + 0) * DMODEL,
                      d_lnb + (long)(i * 3 + 0) * DMODEL);
    run_attention(stream, w, w.hb, w.memb,
                  dc_w + (long)i * WSTRIDE, dc_b + (long)i * 4 * DMODEL,
                  dc_rel + (long)i * RREL * HDIM, /*causal=*/0);
    run_ln(stream, w, d_lng + (long)(i * 3 + 1) * DMODEL,
                      d_lnb + (long)(i * 3 + 1) * DMODEL);
    run_ffn(stream, w, dw1 + (long)i * DMODEL * DFFN, db1 + (long)i * DFFN,
                      dw2 + (long)i * DFFN * DMODEL, db2 + (long)i * DMODEL);
    run_ln(stream, w, d_lng + (long)(i * 3 + 2) * DMODEL,
                      d_lnb + (long)(i * 3 + 2) * DMODEL);
  }

  // h [B,S,D] -> out [S,B,D]
  tr_sbd_kernel<<<eb, 256, 0, stream>>>(w.h, (float*)d_out, 0);
}
